// Attention_38491496907192
// MI455X (gfx1250) — matrix operand order repaired and checked by reading, not compiled, not run on hardware
//
#include <hip/hip_runtime.h>

typedef __attribute__((ext_vector_type(16))) _Float16     v16h;
typedef __attribute__((ext_vector_type(8)))  float        v8f;
typedef __attribute__((ext_vector_type(4)))  unsigned int u32x4;
typedef __attribute__((ext_vector_type(4)))  int          i32x4;

#if defined(__has_builtin)
#if __has_builtin(__builtin_amdgcn_global_load_async_to_lds_b128)
#define HAVE_ASYNC_COPY 1
#endif
#endif
#ifndef HAVE_ASYNC_COPY
#define HAVE_ASYNC_COPY 0
#endif

namespace {

constexpr int kBatch = 2;
constexpr int kSeq   = 2048;
constexpr int kDim   = 1024;
constexpr int kHeads = 8;
constexpr int kDh    = 64;
constexpr int kInner = 512;   // heads * dh
constexpr int kN3    = 1536;  // 3 * inner
constexpr int kRows  = kBatch * kSeq;  // 4096

typedef __attribute__((address_space(1))) i32x4 as1_i32x4;
typedef __attribute__((address_space(3))) i32x4 as3_i32x4;

union Frag {
  v16h  v;
  u32x4 u[2];
};

__device__ __forceinline__ v8f wmma_f16(v16h a, v16h b, v8f c) {
  return __builtin_amdgcn_wmma_f32_16x16x32_f16(false, a, false, b, (short)0, c,
                                                false, false);
}

// Copy 16 bytes global -> LDS.  Uses the CDNA5 async-copy path (ASYNCcnt)
// when the builtin exists; otherwise a plain load + ds_store.
__device__ __forceinline__ void copy16_g2l(_Float16* l, const _Float16* g) {
#if HAVE_ASYNC_COPY
  __builtin_amdgcn_global_load_async_to_lds_b128((as1_i32x4*)g, (as3_i32x4*)l,
                                                 0, 0);
#else
  *(u32x4*)l = *(const u32x4*)g;
#endif
}

__device__ __forceinline__ void async_wait0() {
#if HAVE_ASYNC_COPY
#if __has_builtin(__builtin_amdgcn_s_wait_asynccnt)
  __builtin_amdgcn_s_wait_asynccnt(0);
#else
  asm volatile("s_wait_asynccnt 0x0" ::: "memory");
#endif
#endif
}

// A fragment (16x32 f16). p = &A[m_row][k0] for this lane's row (m = lane&15).
__device__ __forceinline__ v16h load_a_frag(const _Float16* p, int half) {
  Frag f;
  f.u[0] = *(const u32x4*)(p + 16 * half);
  f.u[1] = *(const u32x4*)(p + 16 * half + 8);
  return f.v;
}

// B fragment (32x16 f16). p already resolved to this lane's 16 contiguous
// K-halves (k = j + 16*half, n = lane&15).
__device__ __forceinline__ v16h load_b_frag(const _Float16* p) {
  Frag f;
  f.u[0] = *(const u32x4*)(p);
  f.u[1] = *(const u32x4*)(p + 8);
  return f.v;
}

// ---------------------------------------------------------------- LayerNorm
__global__ void ln_kernel(const float* __restrict__ x,
                          const float* __restrict__ w,
                          const float* __restrict__ bz,
                          _Float16* __restrict__ xn) {
  const int row = blockIdx.x;
  const int tid = threadIdx.x;
  const float* xr = x + (size_t)row * kDim;

  float v[4];
  float s = 0.f;
#pragma unroll
  for (int k = 0; k < 4; ++k) {
    v[k] = xr[tid + k * 256];
    s += v[k];
  }
  __shared__ float red[256];
  red[tid] = s;
  __syncthreads();
  for (int off = 128; off > 0; off >>= 1) {
    if (tid < off) red[tid] += red[tid + off];
    __syncthreads();
  }
  const float mu = red[0] * (1.f / kDim);
  __syncthreads();

  s = 0.f;
#pragma unroll
  for (int k = 0; k < 4; ++k) {
    const float d = v[k] - mu;
    s += d * d;
  }
  red[tid] = s;
  __syncthreads();
  for (int off = 128; off > 0; off >>= 1) {
    if (tid < off) red[tid] += red[tid + off];
    __syncthreads();
  }
  const float rs = rsqrtf(red[0] * (1.f / kDim) + 1e-5f);

#pragma unroll
  for (int k = 0; k < 4; ++k) {
    const int c = tid + k * 256;
    xn[(size_t)row * kDim + c] = (_Float16)((v[k] - mu) * rs * w[c] + bz[c]);
  }
}

// ------------------------------------------- fp32 -> f16 convert + transpose
__global__ void cvt_t_kernel(const float* __restrict__ in,
                             _Float16* __restrict__ out, int R, int C) {
  const size_t i = (size_t)blockIdx.x * blockDim.x + threadIdx.x;
  if (i >= (size_t)R * C) return;
  const int r = (int)(i / C);
  const int c = (int)(i % C);
  out[(size_t)c * R + r] = (_Float16)in[i];
}

// --------------------------------------------------------------- QKV GEMM
// C(4096x1536) = xn(4096x1024) * Wqkv ; Bt = Wqkv^T [1536][1024].
// 4 waves/block; block tile 128x64; B tile (32x64) async-staged in LDS,
// double-buffered.  Epilogue scatters to head-major q/k and transposed v.
__global__ __launch_bounds__(128) void qkv_gemm_kernel(
    const _Float16* __restrict__ A, const _Float16* __restrict__ Bt,
    _Float16* __restrict__ qb, _Float16* __restrict__ kb,
    _Float16* __restrict__ vt) {
  const int tid  = threadIdx.x;
  const int wave = tid >> 5;
  const int lane = tid & 31;
  const int half = lane >> 4, l16 = lane & 15;
  const int gm0 = blockIdx.x * 128 + wave * 32;
  const int jb  = blockIdx.y * 64;

  __shared__ __align__(64) _Float16 Bs[2][64][32];

  auto stage = [&](int buf, int k0) {
#pragma unroll
    for (int c = 0; c < 2; ++c) {
      const int chunk = tid + c * 128;      // 256 chunks of 16B = 4 KB
      const int row = chunk >> 2;           // 0..63 (N)
      const int sub = chunk & 3;            // 4 chunks per 64B row
      copy16_g2l(&Bs[buf][row][sub * 8],
                 Bt + (size_t)(jb + row) * kDim + k0 + sub * 8);
    }
  };

  v8f acc[2][4] = {};
  const _Float16* a0 = A + (size_t)(gm0 + l16) * kDim;
  const _Float16* a1 = a0 + (size_t)16 * kDim;

  stage(0, 0);
  constexpr int kIters = kDim / 32;
  for (int kk = 0; kk < kIters; ++kk) {
    const int k0 = kk * 32;
    const int buf = kk & 1;
    async_wait0();
    __syncthreads();
    if (kk + 1 < kIters) stage(buf ^ 1, k0 + 32);
    __builtin_prefetch(a0 + k0 + 64, 0, 3);
    const v16h fa0 = load_a_frag(a0 + k0, half);
    const v16h fa1 = load_a_frag(a1 + k0, half);
#pragma unroll
    for (int t = 0; t < 4; ++t) {
      const v16h fb = load_b_frag(&Bs[buf][t * 16 + l16][16 * half]);
      acc[0][t] = wmma_f16(fa0, fb, acc[0][t]);
      acc[1][t] = wmma_f16(fa1, fb, acc[1][t]);
    }
    __syncthreads();
  }

#pragma unroll
  for (int r = 0; r < 2; ++r) {
#pragma unroll
    for (int t = 0; t < 4; ++t) {
#pragma unroll
      for (int i = 0; i < 8; ++i) {
        const int gm = gm0 + r * 16 + i + 8 * half;
        const int b = gm >> 11;        // / kSeq
        const int n = gm & (kSeq - 1);
        const int j = jb + t * 16 + l16;
        const _Float16 val = (_Float16)acc[r][t][i];
        if (j < kInner) {
          const int h = j >> 6, d = j & 63;
          qb[((size_t)((b * kHeads + h) * kSeq + n)) * kDh + d] = val;
        } else if (j < 2 * kInner) {
          const int j2 = j - kInner;
          const int h = j2 >> 6, d = j2 & 63;
          kb[((size_t)((b * kHeads + h) * kSeq + n)) * kDh + d] = val;
        } else {
          const int j2 = j - 2 * kInner;
          const int h = j2 >> 6, d = j2 & 63;
          vt[((size_t)((b * kHeads + h) * kDh + d)) * kSeq + n] = val;  // V^T
        }
      }
    }
  }
}

// ------------------------------------------------- l2-normalize q (x8) and k
__global__ void qknorm_kernel(_Float16* __restrict__ q,
                              _Float16* __restrict__ k) {
  const int wid = threadIdx.x >> 5, lane = threadIdx.x & 31;
  const size_t row = (size_t)blockIdx.x * 8 + wid;  // covers 2*8*2048 exactly

  _Float16* qr = q + row * kDh;
  float a = (float)qr[lane], b = (float)qr[lane + 32];
  float s = a * a + b * b;
  for (int off = 16; off >= 1; off >>= 1) s += __shfl_xor(s, off, 32);
  float sc = 8.0f * rsqrtf(s + 1e-12f);  // fold cosine-sim scale into q
  qr[lane] = (_Float16)(a * sc);
  qr[lane + 32] = (_Float16)(b * sc);

  _Float16* kr = k + row * kDh;
  a = (float)kr[lane];
  b = (float)kr[lane + 32];
  s = a * a + b * b;
  for (int off = 16; off >= 1; off >>= 1) s += __shfl_xor(s, off, 32);
  sc = rsqrtf(s + 1e-12f);
  kr[lane] = (_Float16)(a * sc);
  kr[lane + 32] = (_Float16)(b * sc);
}

// ------------------------------------------------ flash attention (causal)
// 4 waves/block; block owns 64 query rows (16 per wave) of one (b,h).
// K (32x64) and V^T (64x32) tiles async-staged in double-buffered LDS and
// shared by all 4 waves.  Online softmax in the f32 C layout.
__global__ __launch_bounds__(128) void attn_kernel(
    const _Float16* __restrict__ q, const _Float16* __restrict__ k,
    const _Float16* __restrict__ vt, _Float16* __restrict__ o) {
  const int tid  = threadIdx.x;
  const int wave = tid >> 5;
  const int lane = tid & 31;
  const int half = lane >> 4, l16 = lane & 15;
  const int q0b = blockIdx.x * 64;
  const int q0  = q0b + wave * 16;
  const int bh = blockIdx.y;
  const int b = bh >> 3, h = bh & 7;

  const _Float16* qbase = q + (size_t)bh * kSeq * kDh;
  const _Float16* kbase = k + (size_t)bh * kSeq * kDh;
  const _Float16* vtb   = vt + (size_t)bh * kDh * kSeq;

  __shared__ __align__(64) _Float16 Ks[2][32][64];  // [key][d]
  __shared__ __align__(64) _Float16 Vs[2][64][32];  // [d][key]
  __shared__ __align__(64) _Float16 Ps[4][16][32];  // per-wave P staging

  auto stage = [&](int buf, int j0) {
#pragma unroll
    for (int c = 0; c < 2; ++c) {
      const int chunk = tid + c * 128;  // 256 chunks per tile
      {  // K tile: 32 rows x 128 B
        const int row = chunk >> 3, sub = chunk & 7;
        copy16_g2l(&Ks[buf][row][sub * 8],
                   kbase + (size_t)(j0 + row) * kDh + sub * 8);
      }
      {  // V^T tile: 64 rows x 64 B
        const int row = chunk >> 2, sub = chunk & 3;
        copy16_g2l(&Vs[buf][row][sub * 8],
                   vtb + (size_t)row * kSeq + j0 + sub * 8);
      }
    }
  };

  // Loop-invariant Q fragments (d-chunks 0 and 32 of this wave's 16 rows).
  const _Float16* qrow = qbase + (size_t)(q0 + l16) * kDh;
  const v16h aq0 = load_a_frag(qrow, half);
  const v16h aq1 = load_a_frag(qrow + 32, half);

  v8f oacc[4] = {};
  float M[8], L[8];
#pragma unroll
  for (int i = 0; i < 8; ++i) {
    M[i] = -1e30f;
    L[i] = 0.f;
  }

  const int nblk = (q0b + 64) / 32;  // uniform across the block
  stage(0, 0);
  for (int blk = 0; blk < nblk; ++blk) {
    const int j0 = blk * 32;
    const int buf = blk & 1;
    async_wait0();
    __syncthreads();
    if (blk + 1 < nblk) stage(buf ^ 1, j0 + 32);

    // S = Q * K^T from LDS: two n-tiles x two d-chunks.
    v8f s0 = {}, s1 = {};
    s0 = wmma_f16(aq0, load_b_frag(&Ks[buf][l16][16 * half]), s0);
    s0 = wmma_f16(aq1, load_b_frag(&Ks[buf][l16][32 + 16 * half]), s0);
    s1 = wmma_f16(aq0, load_b_frag(&Ks[buf][16 + l16][16 * half]), s1);
    s1 = wmma_f16(aq1, load_b_frag(&Ks[buf][16 + l16][32 + 16 * half]), s1);

    // causal mask in C layout: row m = i + 8*half, col = l16 (+16).
#pragma unroll
    for (int i = 0; i < 8; ++i) {
      const int qg = q0 + i + 8 * half;
      if (j0 + l16 > qg) s0[i] = -1e30f;
      if (j0 + 16 + l16 > qg) s1[i] = -1e30f;
    }

    // online softmax: per-row reductions live in 16-lane halves.
#pragma unroll
    for (int i = 0; i < 8; ++i) {
      float rm = fmaxf(s0[i], s1[i]);
      rm = fmaxf(rm, __shfl_xor(rm, 1, 32));
      rm = fmaxf(rm, __shfl_xor(rm, 2, 32));
      rm = fmaxf(rm, __shfl_xor(rm, 4, 32));
      rm = fmaxf(rm, __shfl_xor(rm, 8, 32));
      const float mn = fmaxf(M[i], rm);
      const float alpha = __expf(M[i] - mn);
      M[i] = mn;
      const float p0 = __expf(s0[i] - mn);
      const float p1 = __expf(s1[i] - mn);
      float ls = p0 + p1;
      ls += __shfl_xor(ls, 1, 32);
      ls += __shfl_xor(ls, 2, 32);
      ls += __shfl_xor(ls, 4, 32);
      ls += __shfl_xor(ls, 8, 32);
      L[i] = L[i] * alpha + ls;
#pragma unroll
      for (int t = 0; t < 4; ++t) oacc[t][i] = oacc[t][i] * alpha;
      const int m = i + 8 * half;
      Ps[wave][m][l16] = (_Float16)p0;
      Ps[wave][m][16 + l16] = (_Float16)p1;
    }

    // O += P(16x32) * V(32x64); P is wave-private, DS stays in-order per wave.
    const v16h pa = load_a_frag(&Ps[wave][l16][0], half);
#pragma unroll
    for (int t = 0; t < 4; ++t) {
      oacc[t] =
          wmma_f16(pa, load_b_frag(&Vs[buf][t * 16 + l16][16 * half]), oacc[t]);
    }
    __syncthreads();
  }

  // finalize: divide by L, emit o in [b][n][h*64+d] layout for the out GEMM.
#pragma unroll
  for (int i = 0; i < 8; ++i) {
    const float inv = 1.0f / L[i];
    const int m = i + 8 * half;
    const size_t orow = ((size_t)b * kSeq + q0 + m) * kInner + h * kDh;
#pragma unroll
    for (int t = 0; t < 4; ++t) {
      o[orow + t * 16 + l16] = (_Float16)(oacc[t][i] * inv);
    }
  }
}

// --------------------------------------------------------------- out GEMM
// out(4096x1024) = o(4096x512) * Wout ; Bt = Wout^T [1024][512].
__global__ __launch_bounds__(128) void out_gemm_kernel(
    const _Float16* __restrict__ A, const _Float16* __restrict__ Bt,
    float* __restrict__ out) {
  const int tid  = threadIdx.x;
  const int wave = tid >> 5;
  const int lane = tid & 31;
  const int half = lane >> 4, l16 = lane & 15;
  const int gm0 = blockIdx.x * 128 + wave * 32;
  const int jb  = blockIdx.y * 64;

  __shared__ __align__(64) _Float16 Bs[2][64][32];

  auto stage = [&](int buf, int k0) {
#pragma unroll
    for (int c = 0; c < 2; ++c) {
      const int chunk = tid + c * 128;
      const int row = chunk >> 2;
      const int sub = chunk & 3;
      copy16_g2l(&Bs[buf][row][sub * 8],
                 Bt + (size_t)(jb + row) * kInner + k0 + sub * 8);
    }
  };

  v8f acc[2][4] = {};
  const _Float16* a0 = A + (size_t)(gm0 + l16) * kInner;
  const _Float16* a1 = a0 + (size_t)16 * kInner;

  stage(0, 0);
  constexpr int kIters = kInner / 32;
  for (int kk = 0; kk < kIters; ++kk) {
    const int k0 = kk * 32;
    const int buf = kk & 1;
    async_wait0();
    __syncthreads();
    if (kk + 1 < kIters) stage(buf ^ 1, k0 + 32);
    __builtin_prefetch(a0 + k0 + 64, 0, 3);
    const v16h fa0 = load_a_frag(a0 + k0, half);
    const v16h fa1 = load_a_frag(a1 + k0, half);
#pragma unroll
    for (int t = 0; t < 4; ++t) {
      const v16h fb = load_b_frag(&Bs[buf][t * 16 + l16][16 * half]);
      acc[0][t] = wmma_f16(fa0, fb, acc[0][t]);
      acc[1][t] = wmma_f16(fa1, fb, acc[1][t]);
    }
    __syncthreads();
  }

#pragma unroll
  for (int r = 0; r < 2; ++r) {
#pragma unroll
    for (int t = 0; t < 4; ++t) {
#pragma unroll
      for (int i = 0; i < 8; ++i) {
        const int gm = gm0 + r * 16 + i + 8 * half;
        out[(size_t)gm * kDim + jb + t * 16 + l16] = acc[r][t][i];
      }
    }
  }
}

}  // namespace

extern "C" void kernel_launch(void* const* d_in, const int* in_sizes, int n_in,
                              void* d_out, int out_size, void* d_ws,
                              size_t ws_size, hipStream_t stream) {
  const float* x    = (const float*)d_in[0];
  const float* ln_w = (const float*)d_in[1];
  const float* ln_b = (const float*)d_in[2];
  const float* wqkv = (const float*)d_in[3];
  const float* wout = (const float*)d_in[4];
  float* out = (float*)d_out;

  // Workspace layout (f16), ~28 MB total.
  _Float16* xn     = (_Float16*)d_ws;                          // 4096 x 1024
  _Float16* wqkv_t = xn + (size_t)kRows * kDim;                // 1536 x 1024
  _Float16* wout_t = wqkv_t + (size_t)kN3 * kDim;              // 1024 x 512
  _Float16* qb     = wout_t + (size_t)kDim * kInner;           // [bh][n][64]
  _Float16* kb     = qb + (size_t)kBatch * kHeads * kSeq * kDh;
  _Float16* vt     = kb + (size_t)kBatch * kHeads * kSeq * kDh;  // [bh][64][n]
  _Float16* ob     = vt + (size_t)kBatch * kHeads * kDh * kSeq;  // 4096 x 512

  ln_kernel<<<kRows, 256, 0, stream>>>(x, ln_w, ln_b, xn);
  cvt_t_kernel<<<(kDim * kN3 + 255) / 256, 256, 0, stream>>>(wqkv, wqkv_t,
                                                             kDim, kN3);
  cvt_t_kernel<<<(kInner * kDim + 255) / 256, 256, 0, stream>>>(wout, wout_t,
                                                                kInner, kDim);
  qkv_gemm_kernel<<<dim3(kRows / 128, kN3 / 64), 128, 0, stream>>>(
      xn, wqkv_t, qb, kb, vt);
  qknorm_kernel<<<(kBatch * kHeads * kSeq) / 8, 256, 0, stream>>>(qb, kb);
  attn_kernel<<<dim3(kSeq / 64, kBatch * kHeads), 128, 0, stream>>>(qb, kb, vt,
                                                                    ob);
  out_gemm_kernel<<<dim3(kRows / 128, kDim / 64), 128, 0, stream>>>(ob, wout_t,
                                                                    out);
}
